// FCSNN3_53841710023247
// MI455X (gfx1250) — matrix operand order repaired and checked by reading, not compiled, not run on hardware
//
#include <hip/hip_runtime.h>

// Sizes from the reference
#define T_   128
#define B_   512
#define IN_  784
#define H_   2048
#define OUT_ 10

#define KP   800   // K padded to 25*32 for bf16 WMMA k-steps
#define BT   64    // batch tile per block
#define HT   64    // hidden tile per block
#define XSTR 808   // LDS/global row stride (u16); 1616B -> 16B aligned, conflict-free b128 frags
#define WSTR 808

#define TILE_ELEMS (BT * XSTR)   // 51712 u16 per x tile (contiguous in xbf and LDS)

typedef __bf16 bf16_t;
typedef __attribute__((ext_vector_type(16))) bf16_t v16bf;
typedef __attribute__((ext_vector_type(8)))  float  v8f;

union ABFrag { int4 q[2]; v16bf v; };

// fp32 -> bf16 (RNE) via hardware convert, packed pair
__device__ __forceinline__ unsigned int pk_bf16(float lo, float hi) {
  const unsigned short a = __builtin_bit_cast(unsigned short, (bf16_t)lo);
  const unsigned short b = __builtin_bit_cast(unsigned short, (bf16_t)hi);
  return (unsigned int)a | ((unsigned int)b << 16);
}

// ---------------- Tensor Data Mover: one contiguous 103,424 B tile -> LDS ----------------
#if __has_builtin(__builtin_amdgcn_tensor_load_to_lds)
#define HAVE_TDM 1
typedef unsigned int v4u __attribute__((ext_vector_type(4)));
typedef int          v8i __attribute__((ext_vector_type(8)));
typedef int          v4i __attribute__((ext_vector_type(4)));

__device__ __forceinline__ void tdm_load_tile(const unsigned short* gsrc, unsigned ldsAddr) {
  const unsigned long long ga = (unsigned long long)gsrc;  // global byte address
  // D# group 0: count=1 | lds_addr | global_addr[56:0] | type=2  (CDNA5 ISA)
  const v4u g0 = { 1u, ldsAddr, (unsigned)ga,
                   (unsigned)((ga >> 32) & 0x01FFFFFFull) | (2u << 30) };
  // D# group 1 (8.4): wg_mask=0, data_size=2B; tensor_dim0=tile_dim0=51712 elems,
  // tensor_dim1=tile_dim1=1, stride0=51712; no pad/iterate/barrier.
  const v8i g1 = { 0x00010000,                      // data_size=1 (2 bytes)
                   (int)((TILE_ELEMS & 0xFFFF) << 16),          // tensor_dim0[15:0]
                   (int)((TILE_ELEMS >> 16) | (1 << 16)),       // tensor_dim0[31:16] | tensor_dim1[15:0]
                   (int)((TILE_ELEMS & 0xFFFF) << 16),          // tensor_dim1[31:16]=0 | tile_dim0
                   0x00000001,                      // tile_dim1=1, tile_dim2=0
                   TILE_ELEMS,                      // tensor_dim0_stride[31:0]
                   0, 0 };                          // stride0[47:32], stride1
  const v4i gz = { 0, 0, 0, 0 };
#if defined(__clang_major__) && (__clang_major__ >= 23)
  const v8i gz8 = { 0, 0, 0, 0, 0, 0, 0, 0 };
  __builtin_amdgcn_tensor_load_to_lds(g0, g1, gz, gz, gz8, 0);
#else
  __builtin_amdgcn_tensor_load_to_lds(g0, g1, gz, gz, 0);
#endif
}
#endif

// One-shot x fp32 -> bf16, K zero-padded to stride 808 so each (t,b-tile) is one
// contiguous TDM-able run (bandwidth-bound: 205 MB rd + 106 MB wr ~= 14 us @ 23.3 TB/s)
__global__ __launch_bounds__(256) void xcvt_kernel(const float* __restrict__ x,
                                                   unsigned short* __restrict__ xbf) {
  const int QPR = XSTR / 4;                       // 202 quads per (t,b) row
  const int q   = blockIdx.x * 256 + threadIdx.x; // quad index over T*B*QPR
  const int row = q / QPR;                        // t*B + b
  const int c4  = (q % QPR) * 4;
  uint2 p;
  if (c4 < IN_) {
    const float4 f = *(const float4*)(x + (size_t)row * IN_ + c4);
    p.x = pk_bf16(f.x, f.y);
    p.y = pk_bf16(f.z, f.w);
  } else { p.x = 0u; p.y = 0u; }
  *(uint2*)(xbf + (size_t)row * XSTR + c4) = p;
}

// Fused scan: per (64b x 64h) tile, iterate all T timesteps.
//   cur = x_t @ W1.T via v_wmma_f32_16x16x32_bf16; W1 tile LDS-resident for all T;
//   x_t tile double-buffered in LDS, staged by the Tensor Data Mover (TENSORcnt)
//   so the DMA overlaps the 50 WMMAs of the current step; LIF state in WMMA
//   accumulator layout; S[b,h] = sum_t (0.9^(T-1-t) - 0.8^(T-1-t)) * z_t[b,h].
template <int PRECONV>
__global__ __launch_bounds__(256) void snn_scan_kernel(
    const float* __restrict__ x, const unsigned short* __restrict__ xbf,
    const float* __restrict__ W1, float* __restrict__ Sbuf) {
  __shared__ __align__(16) unsigned short Ws[HT * WSTR];    // ~101 KB, resident all T
  __shared__ __align__(16) unsigned short Xs[2][TILE_ELEMS]; // 2 x ~101 KB double buffer

  const int tid = threadIdx.x;
  const int b0  = blockIdx.x * BT;
  const int h0  = blockIdx.y * HT;

  // ---- stage W1 tile once (fp32 -> bf16, zero-pad K) ----
  for (int j = tid; j < HT * (KP / 4); j += 256) {
    const int row = j / (KP / 4);
    const int c4  = (j % (KP / 4)) * 4;
    uint2 p;
    if (c4 < IN_) {
      const float4 f = *(const float4*)(W1 + (size_t)(h0 + row) * IN_ + c4);
      p.x = pk_bf16(f.x, f.y);
      p.y = pk_bf16(f.z, f.w);
    } else { p.x = 0u; p.y = 0u; }
    *(uint2*)&Ws[row * WSTR + c4] = p;
  }

  // ---- per-wave WMMA tile assignment: 4x4 grid of 16x16 tiles, 2 per wave ----
  const int  wv     = tid >> 5;
  const int  lane   = tid & 31;
  const int  m      = lane & 15;
  const bool laneLo = lane < 16;
  const int  rblk   = wv & 3;          // A row-block (shared by both tiles)
  const int  c0     = (wv >> 2) * 2;   // first B col-block

  // LDS element offsets in contiguous K order (lane half h holds k = 16h + s, same for A and B)
  const int aoff  = (rblk * 16 + m) * XSTR + (laneLo ? 0 : 16);  // A: 16 contiguous K
  const int hrow0 = (c0 * 16 + m) * WSTR;                        // B tile 0
  const int hrow1 = ((c0 + 1) * 16 + m) * WSTR;                  // B tile 1
  const int bko   = laneLo ? 0 : 16;                             // B: 16 contiguous K

  v8f v1a = {}, i1a = {}, Sa = {};
  v8f v1b = {}, i1b = {}, Sb = {};

  // readout weights: w_t = 0.9^(T-1-t) - 0.8^(T-1-t)
  float pa = 1.0f, pb = 1.0f;
  #pragma unroll 1
  for (int k = 0; k < T_ - 1; ++k) { pa *= 0.9f; pb *= 0.8f; }

  // ---- x_t tile staging into buffer `buf` ----
  auto stageX = [&](int t, int buf) {
    if (PRECONV) {
      const unsigned short* xt = xbf + ((size_t)t * B_ + b0) * XSTR;
#ifdef HAVE_TDM
      if (wv == 0)   // one TDM descriptor per tile, issued by wave 0 only
        tdm_load_tile(xt, (unsigned)(unsigned long long)(const void*)&Xs[buf][0]);
#else
      for (int j = tid; j < TILE_ELEMS / 8; j += 256)
        *(uint4*)&Xs[buf][j * 8] = *(const uint4*)(xt + (size_t)j * 8);
#endif
    } else {
      // fused fp32 -> bf16 conversion path
      const float* xt = x + ((size_t)t * B_ + b0) * IN_;
      for (int j = tid; j < BT * (KP / 4); j += 256) {
        const int row = j / (KP / 4);
        const int c4  = (j % (KP / 4)) * 4;
        uint2 p;
        if (c4 < IN_) {
          const float4 f = *(const float4*)(xt + (size_t)row * IN_ + c4);
          p.x = pk_bf16(f.x, f.y);
          p.y = pk_bf16(f.z, f.w);
        } else { p.x = 0u; p.y = 0u; }
        *(uint2*)&Xs[buf][row * XSTR + c4] = p;
      }
    }
  };

  stageX(0, 0);
#ifdef HAVE_TDM
  if (PRECONV && wv == 0) __builtin_amdgcn_s_wait_tensorcnt(0);
#endif
  __syncthreads();

  for (int t = 0; t < T_; ++t) {
    const bool more = (t + 1 < T_);
    // prefetch next timestep into the other buffer; DMA overlaps the WMMAs below
    if (more) stageX(t + 1, (t + 1) & 1);

    const unsigned short* Xc = Xs[t & 1];

    // ---- cur = x_t @ W1.T for this tile: 25 k-steps, 2 WMMAs each ----
    v8f ca = {}, cb = {};
    #pragma unroll 5
    for (int ks = 0; ks < KP / 32; ++ks) {
      const int k0 = ks * 32;
      ABFrag A, Bf0, Bf1;
      const int ao = aoff + k0;
      A.q[0]   = *(const int4*)&Xc[ao];
      A.q[1]   = *(const int4*)&Xc[ao + 8];
      const int bo0 = hrow0 + k0 + bko;
      Bf0.q[0] = *(const int4*)&Ws[bo0];
      Bf0.q[1] = *(const int4*)&Ws[bo0 + 8];
      const int bo1 = hrow1 + k0 + bko;
      Bf1.q[0] = *(const int4*)&Ws[bo1];
      Bf1.q[1] = *(const int4*)&Ws[bo1 + 8];
      ca = __builtin_amdgcn_wmma_f32_16x16x32_bf16(false, A.v, false, Bf0.v,
                                                   (short)0, ca, false, false);
      cb = __builtin_amdgcn_wmma_f32_16x16x32_bf16(false, A.v, false, Bf1.v,
                                                   (short)0, cb, false, false);
    }

    // ---- LIF update (elementwise, accumulator layout) ----
    const float wt = pa - pb;
    #pragma unroll
    for (int j = 0; j < 8; ++j) {
      float v  = 0.9f * v1a[j] + 0.1f * i1a[j];
      float z  = (v > 0.25f) ? 1.0f : 0.0f;
      v1a[j]   = (z != 0.0f) ? 0.0f : v;
      i1a[j]   = 0.8f * i1a[j] + ca[j];
      Sa[j]   += wt * z;

      float v2 = 0.9f * v1b[j] + 0.1f * i1b[j];
      float z2 = (v2 > 0.25f) ? 1.0f : 0.0f;
      v1b[j]   = (z2 != 0.0f) ? 0.0f : v2;
      i1b[j]   = 0.8f * i1b[j] + cb[j];
      Sb[j]   += wt * z2;
    }
    pa *= (1.0f / 0.9f);
    pb *= 1.25f;

#ifdef HAVE_TDM
    if (PRECONV && wv == 0 && more) __builtin_amdgcn_s_wait_tensorcnt(0);
#endif
    __syncthreads();  // buf[t&1] reads done AND buf[(t+1)&1] DMA/stores visible
  }

  // ---- write weighted spike sums (C/D f32 layout: VGPR j -> M = j + (laneHi?8:0)) ----
  #pragma unroll
  for (int j = 0; j < 8; ++j) {
    const int gr  = b0 + rblk * 16 + j + (laneLo ? 0 : 8);
    const int gc0 = h0 + c0 * 16 + m;
    Sbuf[(size_t)gr * H_ + gc0]      = Sa[j];
    Sbuf[(size_t)gr * H_ + gc0 + 16] = Sb[j];
  }
}

// vo[b,o] = sum_h S[b,h] * Wout[o,h]  (tiny: 512x10x2048)
__global__ __launch_bounds__(256) void snn_readout_kernel(
    const float* __restrict__ S, const float* __restrict__ Wout,
    float* __restrict__ vo) {
  const int b   = blockIdx.x;
  const int tid = threadIdx.x;
  float acc[OUT_];
  #pragma unroll
  for (int o = 0; o < OUT_; ++o) acc[o] = 0.0f;

  for (int h = tid; h < H_; h += 256) {
    const float s = S[(size_t)b * H_ + h];
    #pragma unroll
    for (int o = 0; o < OUT_; ++o) acc[o] += s * Wout[o * H_ + h];
  }

  __shared__ float red[256];
  for (int o = 0; o < OUT_; ++o) {
    red[tid] = acc[o];
    __syncthreads();
    for (int s = 128; s > 0; s >>= 1) {
      if (tid < s) red[tid] += red[tid + s];
      __syncthreads();
    }
    if (tid == 0) vo[b * OUT_ + o] = red[0];
    __syncthreads();
  }
}

extern "C" void kernel_launch(void* const* d_in, const int* in_sizes, int n_in,
                              void* d_out, int out_size, void* d_ws, size_t ws_size,
                              hipStream_t stream) {
  const float* x    = (const float*)d_in[0];   // [128,512,784] f32
  const float* W1   = (const float*)d_in[1];   // [2048,784]    f32
  const float* Wout = (const float*)d_in[2];   // [10,2048]     f32
  float*       vo   = (float*)d_out;           // [512,10]      f32

  const size_t s_bytes   = (size_t)B_ * H_ * sizeof(float);                  // 4 MB spike sums
  const size_t xbf_bytes = (size_t)T_ * B_ * XSTR * sizeof(unsigned short);  // ~106 MB bf16 x

  float* S = (float*)d_ws;
  dim3 grid(B_ / BT, H_ / HT);                 // 8 x 32 = 256 blocks

  if (ws_size >= s_bytes + xbf_bytes) {
    // Fast path: pre-convert x to padded bf16 once; scan stages tiles via TDM.
    unsigned short* xbf = (unsigned short*)((char*)d_ws + s_bytes);
    const int quads = T_ * B_ * (XSTR / 4);
    xcvt_kernel<<<quads / 256, 256, 0, stream>>>(x, xbf);
    snn_scan_kernel<1><<<grid, 256, 0, stream>>>(x, xbf, W1, S);
  } else {
    // Fallback: fused fp32 -> bf16 conversion inside the scan.
    snn_scan_kernel<0><<<grid, 256, 0, stream>>>(x, nullptr, W1, S);
  }
  snn_readout_kernel<<<B_, 256, 0, stream>>>(S, Wout, vo);
}
